// EnhancedTextAttentionBlock_62397284876961
// MI455X (gfx1250) — compile-verified, matrix operand order checked by reading, not run on hardware
//
#include <hip/hip_runtime.h>

#define B_DIM 2
#define C_DIM 256
#define S_DIM 2304
#define T_DIM 512
#define NHEAD 8
#define HD 32

typedef __attribute__((ext_vector_type(16))) _Float16 v16h;
typedef __attribute__((ext_vector_type(8)))  _Float16 v8h;
typedef __attribute__((ext_vector_type(8)))  float    v8f;

#define XOR_MAX(v) { v = fmaxf(v, __shfl_xor(v, 8, 32)); v = fmaxf(v, __shfl_xor(v, 4, 32)); \
                     v = fmaxf(v, __shfl_xor(v, 2, 32)); v = fmaxf(v, __shfl_xor(v, 1, 32)); }
#define XOR_SUM(v) { v += __shfl_xor(v, 8, 32); v += __shfl_xor(v, 4, 32); \
                     v += __shfl_xor(v, 2, 32); v += __shfl_xor(v, 1, 32); }

__device__ __forceinline__ v8f wmma16(v16h a, v16h b, v8f c) {
    // D = A(16x32 f16) * B(32x16 f16) + C(16x16 f32)
    return __builtin_amdgcn_wmma_f32_16x16x32_f16(false, a, false, b, (short)0, c, false, false);
}

// Load a 16x32-half fragment whose 16 "rows" (M for A, N for B of an X*W^T GEMM)
// are K-contiguous in memory with row stride ld (elements).
// ISA layout: lane l holds row l&15; elements 0..7 -> K = 8h+i, 8..15 -> K = 16+8h+(i-8), h = l>>4.
__device__ __forceinline__ v16h load_frag_k(const _Float16* p, int ld) {
    int lane = threadIdx.x & 31;
    const _Float16* base = p + (size_t)(lane & 15) * ld + 8 * (lane >> 4);
    v8h lo = *(const v8h*)(base);
    v8h hi = *(const v8h*)(base + 16);
    return __builtin_shufflevector(lo, hi, 0,1,2,3,4,5,6,7,8,9,10,11,12,13,14,15);
}

__device__ __forceinline__ int frag_k_idx(int i, int hg) {
    return (i < 8) ? (8 * hg + i) : (8 + 8 * hg + i); // == 16 + 8h + (i-8) for i>=8
}

__device__ __forceinline__ float block_sum(float v, float* red, int tid) {
    red[tid] = v; __syncthreads();
    for (int o2 = 128; o2 > 0; o2 >>= 1) {
        if (tid < o2) red[tid] += red[tid + o2];
        __syncthreads();
    }
    float r = red[0]; __syncthreads();
    return r;
}

// ---------------- kernel 0: f32 -> f16 weight conversion (q,k,v,o concatenated) ----
__global__ __launch_bounds__(256) void wconv_kernel(const float* qw, const float* kw,
                                                    const float* vw, const float* ow,
                                                    _Float16* dst) {
    int idx = blockIdx.x * 256 + threadIdx.x;        // 0 .. 4*65536-1
    int which = idx >> 16, o = idx & 65535;
    const float* src = (which == 0) ? qw : (which == 1) ? kw : (which == 2) ? vw : ow;
    dst[idx] = (_Float16)src[o];
}

// ---------------- kernel 1: text modulation MLP -> mod[B,C] ------------------------
__global__ __launch_bounds__(256) void textmod_kernel(const float* text, const float* w1,
        const float* b1, const float* g1, const float* bb1, const float* w2,
        const float* b2, const float* g2, const float* bb2, float* mod) {
    int b = blockIdx.x, j = threadIdx.x;
    __shared__ float tf[T_DIM];
    __shared__ float red[256];
    __shared__ float h1[256];
    tf[j] = text[b * T_DIM + j];
    tf[j + 256] = text[b * T_DIM + 256 + j];
    __syncthreads();
    float acc = b1[j];
    for (int t = 0; t < T_DIM; ++t) acc += tf[t] * w1[j * T_DIM + t];
    float sum = block_sum(acc, red, j);
    float sq  = block_sum(acc * acc, red, j);
    float mu = sum * (1.f / 256.f);
    float rstd = rsqrtf(sq * (1.f / 256.f) - mu * mu + 1e-5f);
    float hv = fmaxf((acc - mu) * rstd * g1[j] + bb1[j], 0.f);
    h1[j] = hv;
    __syncthreads();
    float a2 = b2[j];
    for (int t = 0; t < 256; ++t) a2 += h1[t] * w2[j * 256 + t];
    float sum2 = block_sum(a2, red, j);
    float sq2  = block_sum(a2 * a2, red, j);
    float mu2 = sum2 * (1.f / 256.f);
    float rstd2 = rsqrtf(sq2 * (1.f / 256.f) - mu2 * mu2 + 1e-5f);
    float o = (a2 - mu2) * rstd2 * g2[j] + bb2[j];
    mod[b * 256 + j] = 1.f / (1.f + __expf(-o));
}

// ---------------- kernel 2: h-invariant conv -> posrow[B,3,C,48] -------------------
// pe is independent of h, so the 3x3 conv has only 3 distinct output row types.
__global__ __launch_bounds__(256) void posconv_kernel(const float* mod, const float* conv_w,
                                                      const float* conv_b, float* posrow) {
    int bt = blockIdx.x;                 // b*3 + rowtype
    int coTile = blockIdx.y;             // 0..15
    int b = bt / 3, rt = bt % 3;
    int tid = threadIdx.x;
    __shared__ float pre[C_DIM][48];     // 48 KB
    const float LOG_DIV = -9.210340371976184f / 128.0f;  // -ln(10000)/(C/2)
    for (int idx = tid; idx < C_DIM * 48; idx += 256) {
        int ci = idx / 48, w = idx % 48;
        float dv = __expf((float)(ci >> 1) * LOG_DIV);
        float arg = (float)w * dv;
        float pv = (ci & 1) ? __cosf(arg) : __sinf(arg);
        pre[ci][w] = pv * mod[b * 256 + ci];
    }
    __syncthreads();
    int kystart = (rt == 0) ? 1 : 0;
    int kyend   = (rt == 2) ? 1 : 2;
    for (int o = tid; o < 16 * 48; o += 256) {
        int co = coTile * 16 + o / 48;
        int w = o % 48;
        float acc = conv_b[co];
        for (int ci = 0; ci < C_DIM; ++ci) {
            const float* wp = conv_w + ((size_t)co * C_DIM + ci) * 9;
            #pragma unroll
            for (int kx = 0; kx < 3; ++kx) {
                int ww = w + kx - 1;
                if (ww < 0 || ww > 47) continue;
                float wsum = 0.f;
                for (int ky = kystart; ky <= kyend; ++ky) wsum += wp[ky * 3 + kx];
                acc += wsum * pre[ci][ww];
            }
        }
        posrow[(((size_t)b * 3 + rt) * C_DIM + co) * 48 + w] = acc;
    }
}

// ---------------- kernel 3: tokens, LN(q), LN(kv), gate, f16 pack ------------------
__global__ __launch_bounds__(256) void token_kernel(const float* x, const float* posrow,
        const float* nqg, const float* nqb, const float* nkg, const float* nkb,
        const float* gw, _Float16* qn16, _Float16* kn16, float* gate) {
    int bs = blockIdx.x;
    int b = bs / S_DIM, s = bs % S_DIM;
    int h = s / 48, w = s % 48;
    int rt = (h == 0) ? 0 : ((h == 47) ? 2 : 1);
    int c = threadIdx.x;
    __shared__ float red[256];
    float tok = x[((size_t)b * C_DIM + c) * S_DIM + s]
              + posrow[(((size_t)b * 3 + rt) * C_DIM + c) * 48 + w];
    float sum = block_sum(tok, red, c);
    float sq  = block_sum(tok * tok, red, c);
    float mu = sum * (1.f / 256.f);
    float rstd = rsqrtf(sq * (1.f / 256.f) - mu * mu + 1e-5f);
    float xc = (tok - mu) * rstd;
    float qn = xc * nqg[c] + nqb[c];
    float kn = xc * nkg[c] + nkb[c];
    qn16[(size_t)bs * C_DIM + c] = (_Float16)qn;
    kn16[(size_t)bs * C_DIM + c] = (_Float16)kn;
    float gs = block_sum(qn * gw[c], red, c);
    if (c == 0) gate[bs] = 1.f / (1.f + __expf(-gs));
}

// ---------------- kernel 4: WMMA Q/K/V projection ---------------------------------
// One wave computes a 16x64 tile of X @ W^T; z selects Q/K/V. Q pre-scaled by hd^-0.5.
__global__ __launch_bounds__(32) void qkv_kernel(const _Float16* qn16, const _Float16* kn16,
        const _Float16* w16base, const float* qb, const float* kb, const float* vb,
        _Float16* q16, _Float16* k16, _Float16* v16) {
    int which = blockIdx.z;
    const _Float16* act  = (which == 0) ? qn16 : kn16;
    const _Float16* wmat = w16base + (size_t)which * 65536;
    const float* bias    = (which == 0) ? qb : (which == 1) ? kb : vb;
    _Float16* dst        = (which == 0) ? q16 : (which == 1) ? k16 : v16;
    float scale = (which == 0) ? 0.17677669529663689f : 1.f;

    int rowTile = blockIdx.x;            // 0..287 (B*S/16)
    int colTile = blockIdx.y;            // 0..3  -> 64 cols
    int lane = threadIdx.x, r15 = lane & 15, hg = lane >> 4;

    const _Float16* arow = act + (size_t)rowTile * 16 * C_DIM;
    v8f zero = {0.f,0.f,0.f,0.f,0.f,0.f,0.f,0.f};
    v8f acc[4]; acc[0] = zero; acc[1] = zero; acc[2] = zero; acc[3] = zero;
    for (int k0 = 0; k0 < C_DIM; k0 += 32) {
        v16h a = load_frag_k(arow + k0, C_DIM);
        #pragma unroll
        for (int j = 0; j < 4; ++j) {
            v16h bf = load_frag_k(wmat + (size_t)(colTile * 64 + j * 16) * C_DIM + k0, C_DIM);
            acc[j] = wmma16(a, bf, acc[j]);
        }
    }
    int bb = rowTile / 144;
    int sbase = (rowTile % 144) * 16;
    #pragma unroll
    for (int j = 0; j < 4; ++j) {
        int cc = colTile * 64 + j * 16 + r15;
        int head = cc >> 5, d = cc & 31;
        float bval = bias[cc];
        #pragma unroll
        for (int rr = 0; rr < 8; ++rr) {
            int s = sbase + rr + 8 * hg;
            dst[((size_t)(bb * NHEAD + head) * S_DIM + s) * HD + d] =
                (_Float16)((acc[j][rr] + bval) * scale);
        }
    }
}

// ---------------- kernel 5: WMMA flash attention ----------------------------------
// One wave per (b,head, 16-row q-tile); stream 32-key tiles with online softmax.
__global__ __launch_bounds__(32) void attn_kernel(const _Float16* q16, const _Float16* k16,
                                                  const _Float16* v16, _Float16* ao16) {
    int bh = blockIdx.x;                 // b*8 + head
    int qt = blockIdx.y;                 // 0..143
    int b = bh >> 3, head = bh & 7;
    int lane = threadIdx.x;
    int r15 = lane & 15, hg = lane >> 4;

    __shared__ alignas(16) _Float16 vt[32 * HD];   // V tile staged through LDS
    __shared__ alignas(16) _Float16 pt[16 * 32];   // P repack C-layout -> A-layout

    v16h qa = load_frag_k(q16 + ((size_t)bh * S_DIM + qt * 16) * HD, HD);

    v8f zero = {0.f,0.f,0.f,0.f,0.f,0.f,0.f,0.f};
    v8f acc0 = zero, acc1 = zero;
    float mloc[8], lloc[8];
    #pragma unroll
    for (int r = 0; r < 8; ++r) { mloc[r] = -1e30f; lloc[r] = 0.f; }

    for (int kt = 0; kt < S_DIM / 32; ++kt) {
        const _Float16* Kb = k16 + ((size_t)bh * S_DIM + kt * 32) * HD;
        const _Float16* Vb = v16 + ((size_t)bh * S_DIM + kt * 32) * HD;
        __builtin_prefetch(Kb + 32 * HD, 0, 0);    // global_prefetch next K tile
        v16h kb0 = load_frag_k(Kb, HD);            // keys kt*32 .. +15
        v16h kb1 = load_frag_k(Kb + 16 * HD, HD);  // keys +16 .. +31
        {   // stage V tile rows (32 x 32 halfs) into LDS: lane j -> row j
            const v8h* src = (const v8h*)(Vb + lane * HD);
            v8h* dst = (v8h*)(vt + lane * HD);
            dst[0] = src[0]; dst[1] = src[1]; dst[2] = src[2]; dst[3] = src[3];
        }
        v8f s0 = wmma16(qa, kb0, zero);
        v8f s1 = wmma16(qa, kb1, zero);
        #pragma unroll
        for (int r = 0; r < 8; ++r) {
            float mx = fmaxf(s0[r], s1[r]);
            XOR_MAX(mx);                 // row max over 32 keys (16-lane groups share a row)
            float mn = fmaxf(mloc[r], mx);
            float al = __expf(mloc[r] - mn);
            float p0 = __expf(s0[r] - mn);
            float p1 = __expf(s1[r] - mn);
            float rs = p0 + p1;
            XOR_SUM(rs);
            lloc[r] = lloc[r] * al + rs;
            mloc[r] = mn;
            acc0[r] *= al;
            acc1[r] *= al;
            pt[(r + 8 * hg) * 32 + r15] = (_Float16)p0;
            pt[(r + 8 * hg) * 32 + 16 + r15] = (_Float16)p1;
        }
        // P as A-fragment (same wave: DS ops are in-order)
        v16h pa = load_frag_k(pt, 32);
        v16h vb0, vb1;
        #pragma unroll
        for (int i = 0; i < 16; ++i) {
            int kk = frag_k_idx(i, hg);
            vb0[i] = vt[kk * HD + r15];
            vb1[i] = vt[kk * HD + 16 + r15];
        }
        acc0 = wmma16(pa, vb0, acc0);
        acc1 = wmma16(pa, vb1, acc1);
    }
    #pragma unroll
    for (int r = 0; r < 8; ++r) {
        float inv = 1.f / lloc[r];
        size_t row = (size_t)b * S_DIM + qt * 16 + r + 8 * hg;
        _Float16* dst = ao16 + row * C_DIM + head * HD;
        dst[r15]      = (_Float16)(acc0[r] * inv);
        dst[16 + r15] = (_Float16)(acc1[r] * inv);
    }
}

// ---------------- kernel 6: WMMA output proj + gate + LayerNorm + residual ---------
__global__ __launch_bounds__(256) void oproj_kernel(const _Float16* ao16, const _Float16* ow16,
        const float* ob, const float* gate, const float* nog, const float* nob,
        const float* x, float* y) {
    int rowTile = blockIdx.x;            // 0..287
    int b = rowTile / 144;
    int s0 = (rowTile % 144) * 16;
    int tid = threadIdx.x;
    int wave = tid >> 5, lane = tid & 31, r15 = lane & 15, hg = lane >> 4;
    __shared__ float tileo[16][C_DIM];
    __shared__ float gsh[16];

    const _Float16* arow = ao16 + (size_t)rowTile * 16 * C_DIM;
    v8f zero = {0.f,0.f,0.f,0.f,0.f,0.f,0.f,0.f};
    v8f acc0 = zero, acc1 = zero;
    for (int k0 = 0; k0 < C_DIM; k0 += 32) {
        v16h a  = load_frag_k(arow + k0, C_DIM);
        v16h b0 = load_frag_k(ow16 + (size_t)(wave * 32) * C_DIM + k0, C_DIM);
        v16h b1 = load_frag_k(ow16 + (size_t)(wave * 32 + 16) * C_DIM + k0, C_DIM);
        acc0 = wmma16(a, b0, acc0);
        acc1 = wmma16(a, b1, acc1);
    }
    if (tid < 16) gsh[tid] = gate[b * S_DIM + s0 + tid];
    __syncthreads();
    #pragma unroll
    for (int rr = 0; rr < 8; ++rr) {
        int row = rr + 8 * hg;
        float g = gsh[row];
        int c0 = wave * 32 + r15;
        int c1 = wave * 32 + 16 + r15;
        tileo[row][c0] = g * (acc0[rr] + ob[c0]);
        tileo[row][c1] = g * (acc1[rr] + ob[c1]);
    }
    __syncthreads();
    int grp = tid >> 4, j16 = tid & 15;  // 16 threads per row
    float sum = 0.f, sq = 0.f;
    #pragma unroll
    for (int i = 0; i < 16; ++i) {
        float v = tileo[grp][j16 + 16 * i];
        sum += v; sq += v * v;
    }
    XOR_SUM(sum); XOR_SUM(sq);
    float mu = sum * (1.f / 256.f);
    float rstd = rsqrtf(sq * (1.f / 256.f) - mu * mu + 1e-5f);
    int s = s0 + grp;
    #pragma unroll
    for (int i = 0; i < 16; ++i) {
        int c = j16 + 16 * i;
        float v = (tileo[grp][c] - mu) * rstd * nog[c] + nob[c];
        size_t xi = ((size_t)b * C_DIM + c) * S_DIM + s;
        y[xi] = x[xi] + v;
    }
}

// ---------------- launcher ---------------------------------------------------------
extern "C" void kernel_launch(void* const* d_in, const int* in_sizes, int n_in,
                              void* d_out, int out_size, void* d_ws, size_t ws_size,
                              hipStream_t stream) {
    (void)in_sizes; (void)n_in; (void)out_size; (void)ws_size;
    const float* x        = (const float*)d_in[0];
    const float* textf    = (const float*)d_in[1];
    const float* tm_w1    = (const float*)d_in[2];
    const float* tm_b1    = (const float*)d_in[3];
    const float* tm_ln1_g = (const float*)d_in[4];
    const float* tm_ln1_b = (const float*)d_in[5];
    const float* tm_w2    = (const float*)d_in[6];
    const float* tm_b2    = (const float*)d_in[7];
    const float* tm_ln2_g = (const float*)d_in[8];
    const float* tm_ln2_b = (const float*)d_in[9];
    const float* conv_w   = (const float*)d_in[10];
    const float* conv_b   = (const float*)d_in[11];
    const float* q_w      = (const float*)d_in[12];
    const float* q_b      = (const float*)d_in[13];
    const float* k_w      = (const float*)d_in[14];
    const float* k_b      = (const float*)d_in[15];
    const float* v_w      = (const float*)d_in[16];
    const float* v_b      = (const float*)d_in[17];
    const float* o_w      = (const float*)d_in[18];
    const float* o_b      = (const float*)d_in[19];
    const float* gate_w   = (const float*)d_in[20];
    const float* nq_g     = (const float*)d_in[21];
    const float* nq_b     = (const float*)d_in[22];
    const float* nkv_g    = (const float*)d_in[23];
    const float* nkv_b    = (const float*)d_in[24];
    const float* no_g     = (const float*)d_in[25];
    const float* no_b     = (const float*)d_in[26];
    float* out = (float*)d_out;

    char* wsp = (char*)d_ws;
    size_t off = 0;
    auto walloc = [&](size_t bytes) -> void* {
        void* p = wsp + off;
        off += (bytes + 255) & ~(size_t)255;
        return p;
    };
    _Float16* w16    = (_Float16*)walloc((size_t)4 * 65536 * 2);          // q,k,v,o f16
    float*    mod    = (float*)walloc((size_t)B_DIM * C_DIM * 4);
    float*    posrow = (float*)walloc((size_t)B_DIM * 3 * C_DIM * 48 * 4);
    _Float16* qn16   = (_Float16*)walloc((size_t)B_DIM * S_DIM * C_DIM * 2);
    _Float16* kn16   = (_Float16*)walloc((size_t)B_DIM * S_DIM * C_DIM * 2);
    float*    gate   = (float*)walloc((size_t)B_DIM * S_DIM * 4);
    _Float16* q16    = (_Float16*)walloc((size_t)B_DIM * NHEAD * S_DIM * HD * 2);
    _Float16* k16    = (_Float16*)walloc((size_t)B_DIM * NHEAD * S_DIM * HD * 2);
    _Float16* v16    = (_Float16*)walloc((size_t)B_DIM * NHEAD * S_DIM * HD * 2);
    _Float16* ao16   = (_Float16*)walloc((size_t)B_DIM * S_DIM * C_DIM * 2);

    wconv_kernel<<<1024, 256, 0, stream>>>(q_w, k_w, v_w, o_w, w16);
    textmod_kernel<<<B_DIM, 256, 0, stream>>>(textf, tm_w1, tm_b1, tm_ln1_g, tm_ln1_b,
                                              tm_w2, tm_b2, tm_ln2_g, tm_ln2_b, mod);
    posconv_kernel<<<dim3(B_DIM * 3, 16), 256, 0, stream>>>(mod, conv_w, conv_b, posrow);
    token_kernel<<<B_DIM * S_DIM, 256, 0, stream>>>(x, posrow, nq_g, nq_b, nkv_g, nkv_b,
                                                    gate_w, qn16, kn16, gate);
    qkv_kernel<<<dim3(288, 4, 3), 32, 0, stream>>>(qn16, kn16, w16, q_b, k_b, v_b,
                                                   q16, k16, v16);
    attn_kernel<<<dim3(16, 144), 32, 0, stream>>>(q16, k16, v16, ao16);
    oproj_kernel<<<288, 256, 0, stream>>>(ao16, w16 + (size_t)3 * 65536, o_b, gate,
                                          no_g, no_b, x, out);
}
